// Attention_28372553957390
// MI455X (gfx1250) — matrix operand order repaired and checked by reading, not compiled, not run on hardware
//
#include <hip/hip_runtime.h>
#include <hip/hip_bf16.h>
#include <math.h>

// ---------------------------------------------------------------------------
// Additive (Bahdanau) attention for MI455X / gfx1250.
//   B=8, Q=128, K=256, Dq=Dk=Dv=H=512
// Stages:
//   0) one-time fp32->bf16 convert (query,key) and transpose-convert
//      (Wq, Wk, value) so GEMM fragments are contiguous packed bf16
//   1) q_proj = query @ Wq            (WMMA bf16, f32 accum, 16x64/wave)
//   2) k_proj = key   @ Wk            (WMMA bf16, f32 accum)
//   3) scores + mask + softmax        (v_tanh_f32 TRANS kernel, LDS softmax)
//   4) attn_vec = weights @ value     (batched WMMA bf16)
// ---------------------------------------------------------------------------

typedef __attribute__((ext_vector_type(16))) __bf16 v16bf;
typedef __attribute__((ext_vector_type(8)))  float  v8f;

union Frag {
    v16bf v;
    uint4 q[2];
    unsigned short u[16];
};

// fp32 -> bf16 (round-to-nearest-even), returned as raw u16
__device__ __forceinline__ unsigned int f2bf(float x) {
    union { float f; unsigned int u; } c;
    c.f = x;
    unsigned int r = c.u + 0x7FFFu + ((c.u >> 16) & 1u);
    return r >> 16;
}

// gfx1250 hardware tanh (TRANS32 op)
__device__ __forceinline__ float fast_tanh(float x) {
#if __has_builtin(__builtin_amdgcn_tanhf)
    return __builtin_amdgcn_tanhf(x);
#elif __has_builtin(__builtin_amdgcn_tanh_f32)
    return __builtin_amdgcn_tanh_f32(x);
#else
    float y;
    // TRANS co-exec hazard: 1 delay before output use (ISA 7.4/8)
    asm("v_tanh_f32 %0, %1\n\ts_delay_alu instid0(TRANS32_DEP_1)"
        : "=v"(y) : "v"(x));
    return y;
#endif
}

// ---------------------------------------------------------------------------
// Elementwise fp32 -> packed bf16 (4 elems/thread)
// ---------------------------------------------------------------------------
__global__ void __launch_bounds__(256)
convert_bf16_kernel(const float* __restrict__ src,
                    unsigned short* __restrict__ dst, int n4) {
    int i = blockIdx.x * blockDim.x + threadIdx.x;
    if (i >= n4) return;
    float4 v = reinterpret_cast<const float4*>(src)[i];
    uint2 p;
    p.x = f2bf(v.x) | (f2bf(v.y) << 16);
    p.y = f2bf(v.z) | (f2bf(v.w) << 16);
    reinterpret_cast<uint2*>(dst)[i] = p;
}

// ---------------------------------------------------------------------------
// Tiled transpose + convert: src (R x C, f32 row-major) -> dst (C x R, bf16)
// 32x32 LDS tile, blockDim = 256 (32x8), grid (C/32, R/32, batch)
// ---------------------------------------------------------------------------
__global__ void __launch_bounds__(256)
transpose_bf16_kernel(const float* __restrict__ src,
                      unsigned short* __restrict__ dst,
                      int R, int C, long long sStride, long long dStride) {
    __shared__ float tile[32][33];
    src += (long long)blockIdx.z * sStride;
    dst += (long long)blockIdx.z * dStride;
    const int c0 = blockIdx.x * 32, r0 = blockIdx.y * 32;
    const int tx = threadIdx.x & 31, ty = threadIdx.x >> 5;   // 32 x 8

#pragma unroll
    for (int i = 0; i < 32; i += 8)
        tile[ty + i][tx] = src[(size_t)(r0 + ty + i) * C + c0 + tx];
    __syncthreads();
#pragma unroll
    for (int i = 0; i < 32; i += 8)
        dst[(size_t)(c0 + ty + i) * R + r0 + tx] =
            (unsigned short)f2bf(tile[tx][ty + i]);
}

// ---------------------------------------------------------------------------
// WMMA GEMM: C(MxN,f32) = A(MxK,bf16 row-major) * BT(NxK,bf16 row-major)^T
// One wave computes a 16x64 strip (4 accumulators, A fragment reused 4x).
// 4 waves / 128-thread block; gridDim.y = batch.
//
// VGPR layouts per CDNA5 ISA 7.12.2 (wave32):
//   A 16x32: lanes 0-15 -> M=lane,    K slots {k0..k0+7, k0+16..k0+23}
//            lanes 16-31 -> M=lane-16, K slots {k0+8..15, k0+24..31}
//   B 32x16: lanes 0-15 -> N=lane, K slots {k0..k0+7, k0+16..k0+23}; lanes 16-31 -> {k0+8..15, k0+24..31}
//   C 16x16: VGPR r -> row half*8+r, col lane&15
// ---------------------------------------------------------------------------
__global__ void __launch_bounds__(128)
gemm_bf16_tn_kernel(const unsigned short* __restrict__ A, int lda, long long strideA,
                    const unsigned short* __restrict__ BT, int ldb, long long strideB,
                    float* __restrict__ C, int ldc, long long strideC,
                    int M, int N, int Kd) {
    A  += (long long)blockIdx.y * strideA;
    BT += (long long)blockIdx.y * strideB;
    C  += (long long)blockIdx.y * strideC;

    const int lane = threadIdx.x & 31;
    const int half = lane >> 4;
    const int l16  = lane & 15;

    const int n64  = N >> 6;                                  // 64-col strips
    const int wave = blockIdx.x * 4 + (threadIdx.x >> 5);
    const int tm   = wave / n64;
    const int tn   = wave % n64;
    if (tm >= (M >> 4)) return;

    const unsigned short* arow = A + (size_t)(tm * 16 + l16) * lda + half * 8;
    const unsigned short* b0   = BT + (size_t)(tn * 64 + l16) * ldb + half * 8;

    v8f acc0 = {}, acc1 = {}, acc2 = {}, acc3 = {};
    for (int k0 = 0; k0 < Kd; k0 += 32) {
        Frag a;
        a.q[0] = *reinterpret_cast<const uint4*>(arow + k0);
        a.q[1] = *reinterpret_cast<const uint4*>(arow + k0 + 16);

        Frag b;
#pragma unroll
        for (int j = 0; j < 4; ++j) {
            const unsigned short* brow = b0 + (size_t)(j * 16) * ldb + k0;
            b.q[0] = *reinterpret_cast<const uint4*>(brow);
            b.q[1] = *reinterpret_cast<const uint4*>(brow + 16);
            v8f* accp = (j == 0) ? &acc0 : (j == 1) ? &acc1 : (j == 2) ? &acc2 : &acc3;
            *accp = __builtin_amdgcn_wmma_f32_16x16x32_bf16(
                false, a.v, false, b.v, (short)0, *accp, false, false);
        }
    }

    float* crow = C + (size_t)(tm * 16 + half * 8) * ldc + tn * 64 + l16;
#pragma unroll
    for (int j = 0; j < 4; ++j) {
        const v8f acc = (j == 0) ? acc0 : (j == 1) ? acc1 : (j == 2) ? acc2 : acc3;
#pragma unroll
        for (int r = 0; r < 8; ++r)
            crow[(size_t)r * ldc + j * 16] = acc[r];
    }
}

// ---------------------------------------------------------------------------
// scores + mask + scale + softmax.  One 256-thread block per (b,q);
// thread k owns scores[b,q,k].  134M v_tanh_f32 total -> TRANS-bound stage.
// Writes f32 weights to d_out and bf16 weights to workspace.
// ---------------------------------------------------------------------------
__global__ void __launch_bounds__(256)
score_softmax_kernel(const float* __restrict__ qproj,   // (B*Q, H)
                     const float* __restrict__ kproj,   // (B*K, H)
                     const float* __restrict__ b1,      // (H)
                     const float* __restrict__ w2,      // (H)
                     const int*   __restrict__ mask,    // (B, K)
                     float* __restrict__ attw,          // (B*Q, K) f32 out
                     unsigned short* __restrict__ attw_bf) { // (B*Q, K) bf16
    const int H = 512, K = 256, Q = 128;
    const int bq = blockIdx.x;
    const int b  = bq / Q;
    const int k  = threadIdx.x;

    __shared__ __align__(16) float pre[512];
    __shared__ __align__(16) float w2s[512];
    __shared__ float red[256];

    for (int h = k; h < H; h += K) {
        pre[h] = qproj[(size_t)bq * H + h] + b1[h];
        w2s[h] = w2[h];
    }
    __syncthreads();

    const float4* krow4 = reinterpret_cast<const float4*>(kproj + ((size_t)b * K + k) * H);
    const float4* pre4  = reinterpret_cast<const float4*>(pre);
    const float4* w2s4  = reinterpret_cast<const float4*>(w2s);

    float s = 0.f;
#pragma unroll 2
    for (int h4 = 0; h4 < H / 4; ++h4) {
        const float4 kv = krow4[h4];
        const float4 pv = pre4[h4];
        const float4 wv = w2s4[h4];
        s += wv.x * fast_tanh(pv.x + kv.x);
        s += wv.y * fast_tanh(pv.y + kv.y);
        s += wv.z * fast_tanh(pv.z + kv.z);
        s += wv.w * fast_tanh(pv.w + kv.w);
    }

    if (mask[b * K + k] != 0) s = -1.0e9f;
    s *= 0.04419417382415922f;          // 1/sqrt(512)

    // max-reduce
    red[k] = s;
    __syncthreads();
    for (int off = 128; off > 0; off >>= 1) {
        if (k < off) red[k] = fmaxf(red[k], red[k + off]);
        __syncthreads();
    }
    const float mx = red[0];
    __syncthreads();

    // sum-reduce of exp
    const float e = __expf(s - mx);
    red[k] = e;
    __syncthreads();
    for (int off = 128; off > 0; off >>= 1) {
        if (k < off) red[k] += red[k + off];
        __syncthreads();
    }
    const float w = e * (1.0f / red[0]);

    attw[(size_t)bq * K + k]    = w;
    attw_bf[(size_t)bq * K + k] = (unsigned short)f2bf(w);
}

// ---------------------------------------------------------------------------
// Host launcher
// ---------------------------------------------------------------------------
extern "C" void kernel_launch(void* const* d_in, const int* in_sizes, int n_in,
                              void* d_out, int out_size, void* d_ws, size_t ws_size,
                              hipStream_t stream) {
    const int B = 8, Q = 128, K = 256, H = 512, Dq = 512, Dv = 512;

    const float* query = (const float*)d_in[0];   // (B,Q,Dq)
    const float* key   = (const float*)d_in[1];   // (B,K,Dk)
    const float* value = (const float*)d_in[2];   // (B,K,Dv)
    const int*   mask  = (const int*)  d_in[3];   // (B,K)
    const float* W1    = (const float*)d_in[4];   // (Dq+Dk, H)
    const float* b1    = (const float*)d_in[5];   // (H)
    const float* w2    = (const float*)d_in[6];   // (H,1)

    float* attn_vec = (float*)d_out;                          // (B,Q,Dv)
    float* attn_w   = (float*)d_out + (size_t)B * Q * Dv;     // (B,Q,K)

    // ---- workspace layout ----
    char* ws = (char*)d_ws;
    float* qproj = (float*)ws;                     ws += (size_t)B * Q * H * 4;   // 2 MB
    float* kproj = (float*)ws;                     ws += (size_t)B * K * H * 4;   // 4 MB
    unsigned short* q_bf  = (unsigned short*)ws;   ws += (size_t)B * Q * Dq * 2;  // 1 MB
    unsigned short* k_bf  = (unsigned short*)ws;   ws += (size_t)B * K * Dq * 2;  // 2 MB
    unsigned short* WqT   = (unsigned short*)ws;   ws += (size_t)Dq * H * 2;      // 0.5 MB
    unsigned short* WkT   = (unsigned short*)ws;   ws += (size_t)Dq * H * 2;      // 0.5 MB
    unsigned short* valT  = (unsigned short*)ws;   ws += (size_t)B * Dv * K * 2;  // 2 MB
    unsigned short* w_bf  = (unsigned short*)ws;   ws += (size_t)B * Q * K * 2;   // 0.5 MB

    // ---- 0) conversions ----
    {
        int n4 = (B * Q * Dq) / 4;   // query
        convert_bf16_kernel<<<(n4 + 255) / 256, 256, 0, stream>>>(query, q_bf, n4);
        n4 = (B * K * Dq) / 4;       // key
        convert_bf16_kernel<<<(n4 + 255) / 256, 256, 0, stream>>>(key, k_bf, n4);
    }
    // Wq^T, Wk^T : (512x512) -> (512x512)
    {
        dim3 g(H / 32, Dq / 32, 1);
        transpose_bf16_kernel<<<g, 256, 0, stream>>>(W1, WqT, Dq, H, 0, 0);
        transpose_bf16_kernel<<<g, 256, 0, stream>>>(W1 + (size_t)Dq * H, WkT, Dq, H, 0, 0);
    }
    // value^T per batch: (K x Dv) -> (Dv x K)
    {
        dim3 g(Dv / 32, K / 32, B);
        transpose_bf16_kernel<<<g, 256, 0, stream>>>(value, valT, K, Dv,
                                                     (long long)K * Dv, (long long)Dv * K);
    }

    // ---- 1) q_proj: M=1024 N=512 K=512 -> 64*8=512 waves -> 128 blocks ----
    {
        const int M = B * Q, N = H;
        dim3 g(((M >> 4) * (N >> 6)) / 4, 1);
        gemm_bf16_tn_kernel<<<g, 128, 0, stream>>>(q_bf, Dq, 0, WqT, Dq, 0,
                                                   qproj, N, 0, M, N, Dq);
    }
    // ---- 2) k_proj: M=2048 N=512 K=512 -> 1024 waves -> 256 blocks ----
    {
        const int M = B * K, N = H;
        dim3 g(((M >> 4) * (N >> 6)) / 4, 1);
        gemm_bf16_tn_kernel<<<g, 128, 0, stream>>>(k_bf, Dq, 0, WkT, Dq, 0,
                                                   kproj, N, 0, M, N, Dq);
    }
    // ---- 3) scores + softmax ----
    score_softmax_kernel<<<B * Q, 256, 0, stream>>>(qproj, kproj, b1, w2, mask,
                                                    attn_w, w_bf);
    // ---- 4) attn_vec: per batch M=128 N=512 K=256 -> 8*8=64 waves -> 16 blocks ----
    {
        dim3 g(((Q >> 4) * (Dv >> 6)) / 4, B);
        gemm_bf16_tn_kernel<<<g, 128, 0, stream>>>(
            w_bf, K, (long long)Q * K,
            valT, K, (long long)Dv * K,
            attn_vec, Dv, (long long)Q * Dv,
            Q, Dv, K);
    }
}
